// CausalSelfAttention_15822659519015
// MI455X (gfx1250) — matrix operand order repaired and checked by reading, not compiled, not run on hardware
//
#include <hip/hip_runtime.h>

// ---------------------------------------------------------------------------
// CDNA5 (gfx1250) bf16 WMMA causal self-attention.
//   - all matmuls: v_wmma_f32_16x16x32_bf16 (wave32, fp32 accumulate)
//   - GEMM K-loop: ping-pong double buffer, fully unrolled -> no register
//     rotation moves, loads for the next step in flight across current WMMAs
//   - attention K-tile staging: global_load_async_to_lds_b128 (ASYNCcnt path)
// ---------------------------------------------------------------------------

typedef __attribute__((ext_vector_type(8)))  __bf16 v8bf;
typedef __attribute__((ext_vector_type(16))) __bf16 v16bf;
typedef __attribute__((ext_vector_type(8)))  float  v8f;
typedef __attribute__((vector_size(16)))     int    v4i;

__device__ __forceinline__ v8bf ld8(const __bf16* p) { return *(const v8bf*)p; }

__device__ __forceinline__ v16bf cat16(v8bf lo, v8bf hi) {
  return __builtin_shufflevector(lo, hi, 0,1,2,3,4,5,6,7,8,9,10,11,12,13,14,15);
}

__device__ __forceinline__ v8f wmma_bf16(v16bf a, v16bf b, v8f c) {
  return __builtin_amdgcn_wmma_f32_16x16x32_bf16(false, a, false, b, (short)0, c,
                                                 false, false);
}

// Async global -> LDS copy of 16 bytes per lane (ASYNCcnt-tracked).
__device__ __forceinline__ void async_copy16(const __bf16* g, __bf16* l) {
#if __has_builtin(__builtin_amdgcn_global_load_async_to_lds_b128)
  __builtin_amdgcn_global_load_async_to_lds_b128((v4i*)g, (v4i*)l, 0, 0);
#else
  unsigned loff = (unsigned)(unsigned long long)
      (__attribute__((address_space(3))) void*)(void*)l;
  asm volatile("global_load_async_to_lds_b128 %0, %1, off"
               :: "v"(loff), "v"(g) : "memory");
#endif
}

__device__ __forceinline__ void wait_async0() {
#if __has_builtin(__builtin_amdgcn_s_wait_asynccnt)
  __builtin_amdgcn_s_wait_asynccnt(0);
#else
  asm volatile("s_wait_asynccnt 0x0" ::: "memory");
#endif
}

// A-fragment (16x32 bf16): lanes 0-15 -> K 0..15, lanes 16-31 -> K 16..31 (contiguous)
// B-fragment (32x16 bf16): lanes 0-15 -> K 0..15, lanes 16-31 -> K 16..31 (contiguous)

// ---------------------------------------------------------------------------
// Conversion kernels
// ---------------------------------------------------------------------------
__global__ void conv_f32_to_bf16(const float* __restrict__ src,
                                 __bf16* __restrict__ dst, int n) {
  int i = blockIdx.x * 256 + threadIdx.x;
  if (i < n) dst[i] = (__bf16)src[i];
}

// src [R][C] row-major f32 -> dst [C][R] row-major bf16 (transpose)
__global__ void conv_transpose_bf16(const float* __restrict__ src,
                                    __bf16* __restrict__ dst, int R, int C) {
  int i = blockIdx.x * 256 + threadIdx.x;
  if (i < R * C) {
    int r = i / C, c = i % C;
    dst[(size_t)c * R + r] = (__bf16)src[i];
  }
}

// ---------------------------------------------------------------------------
// Ping-pong pipelined per-wave GEMM accumulation:
//   A  [M,K] bf16 row-major, BT [N,K] bf16 row-major (B pre-transposed)
//   wave computes rows [m0,m0+16), cols [n0,n0+64) into acc[4] (n-subtiles)
//   Two fixed fragment buffers alternate load/consume; fully unrolled (K
//   constant at call sites) so there are no rotation moves or branches.
// ---------------------------------------------------------------------------
__device__ __forceinline__ void gemm_wave_acc(const __bf16* __restrict__ A,
                                              const __bf16* __restrict__ BT,
                                              const int K, int m0, int n0,
                                              int lane, v8f acc[4]) {
  const int hi    = lane >> 4;
  const int lnib  = lane & 15;
  const int akoff = hi ? 16 : 0;
  const int bkoff = hi ? 16 : 0;
  const __bf16* arow = A + (size_t)(m0 + lnib) * K + akoff;
  const __bf16* brow[4];
#pragma unroll
  for (int j = 0; j < 4; ++j)
    brow[j] = BT + (size_t)(n0 + j * 16 + lnib) * K + bkoff;

  v8bf al0, ah0, bl0[4], bh0[4];  // buffer 0
  v8bf al1, ah1, bl1[4], bh1[4];  // buffer 1

  // prologue: buffer 0 <- k = 0
  al0 = ld8(arow);
  ah0 = ld8(arow + 8);
#pragma unroll
  for (int j = 0; j < 4; ++j) { bl0[j] = ld8(brow[j]); bh0[j] = ld8(brow[j] + 8); }

#pragma unroll
  for (int k0 = 0; k0 < K; k0 += 64) {
    const int k1 = k0 + 32, k2 = k0 + 64;
    // prefetch buffer 1 <- k1 (in flight across buffer-0 WMMAs)
    if (k1 < K) {
      al1 = ld8(arow + k1);
      ah1 = ld8(arow + k1 + 8);
#pragma unroll
      for (int j = 0; j < 4; ++j) {
        bl1[j] = ld8(brow[j] + k1);
        bh1[j] = ld8(brow[j] + k1 + 8);
      }
    }
    {
      v16bf a = cat16(al0, ah0);
#pragma unroll
      for (int j = 0; j < 4; ++j)
        acc[j] = wmma_bf16(a, cat16(bl0[j], bh0[j]), acc[j]);
    }
    // prefetch buffer 0 <- k2 (in flight across buffer-1 WMMAs)
    if (k2 < K) {
      al0 = ld8(arow + k2);
      ah0 = ld8(arow + k2 + 8);
#pragma unroll
      for (int j = 0; j < 4; ++j) {
        bl0[j] = ld8(brow[j] + k2);
        bh0[j] = ld8(brow[j] + k2 + 8);
      }
    }
    if (k1 < K) {
      v16bf a = cat16(al1, ah1);
#pragma unroll
      for (int j = 0; j < 4; ++j)
        acc[j] = wmma_bf16(a, cat16(bl1[j], bh1[j]), acc[j]);
    }
  }
}

// ---------------------------------------------------------------------------
// QKV projection: xb[8192,512] @ WqkvT -> scatter into Q/K/V [B*H, T, D] bf16
// grid (8192/128, 1536/64), block 256 (8 waves, 16 rows each)
// ---------------------------------------------------------------------------
__global__ __launch_bounds__(256) void qkv_gemm_kernel(
    const __bf16* __restrict__ xb, const __bf16* __restrict__ wqkvT,
    __bf16* __restrict__ qb, __bf16* __restrict__ kb, __bf16* __restrict__ vb) {
  const int lane = threadIdx.x & 31, wid = threadIdx.x >> 5;
  const int m0 = blockIdx.x * 128 + wid * 16;
  const int n0 = blockIdx.y * 64;
  const v8f zero = {0.f, 0.f, 0.f, 0.f, 0.f, 0.f, 0.f, 0.f};
  v8f acc[4] = {zero, zero, zero, zero};
  gemm_wave_acc(xb, wqkvT, 512, m0, n0, lane, acc);

  const int hi = lane >> 4, lnib = lane & 15;
#pragma unroll
  for (int j = 0; j < 4; ++j) {
#pragma unroll
    for (int r = 0; r < 8; ++r) {
      int m = m0 + r + 8 * hi;            // token index in [0,8192)
      int n = n0 + j * 16 + lnib;         // col in [0,1536)
      int b = m >> 11, t = m & 2047;
      int which = n >> 9, c = n & 511;
      int h = c >> 6, d = c & 63;
      __bf16* dst = (which == 0) ? qb : (which == 1) ? kb : vb;
      dst[(((size_t)(b * 8 + h)) * 2048 + t) * 64 + d] = (__bf16)acc[j][r];
    }
  }
}

// ---------------------------------------------------------------------------
// Flash attention (causal, online softmax).
// grid (B*H=32, T/128=16), block 256. Each wave: 16 query rows, full D=64.
// K tile: async global->LDS DMA. V tile: register transpose into LDS.
// ---------------------------------------------------------------------------
__global__ __launch_bounds__(256) void attn_kernel(
    const __bf16* __restrict__ qb, const __bf16* __restrict__ kb,
    const __bf16* __restrict__ vb, __bf16* __restrict__ ao) {
  __shared__ __bf16 Klds[32][64];     // [key][d]
  __shared__ __bf16 Vtlds[64][32];    // [d][key]
  __shared__ __bf16 Plds[8][16][32];  // per-wave P tile [row][key]

  const int lane = threadIdx.x & 31, wid = threadIdx.x >> 5;
  const int hi = lane >> 4, lnib = lane & 15;
  const int bh = blockIdx.x;
  const int q0 = blockIdx.y * 128;
  const __bf16* Q  = qb + (size_t)bh * 2048 * 64;
  const __bf16* Kp = kb + (size_t)bh * 2048 * 64;
  const __bf16* Vp = vb + (size_t)bh * 2048 * 64;

  // Preload this wave's Q fragments (A layout), rows q0+wid*16 .. +15
  const int akoff = hi ? 16 : 0;
  const int bkoff = hi ? 16 : 0;
  const __bf16* qrow = Q + (size_t)(q0 + wid * 16 + lnib) * 64;
  v16bf qa0 = cat16(ld8(qrow + akoff),      ld8(qrow + akoff + 8));
  v16bf qa1 = cat16(ld8(qrow + 32 + akoff), ld8(qrow + 32 + akoff + 8));

  const v8f zero = {0.f, 0.f, 0.f, 0.f, 0.f, 0.f, 0.f, 0.f};
  v8f o0 = zero, o1 = zero, o2 = zero, o3 = zero;
  float mrow[8], lrow[8];
#pragma unroll
  for (int r = 0; r < 8; ++r) { mrow[r] = -1e30f; lrow[r] = 0.f; }

  const float scale2 = 0.125f * 1.44269504088896340736f;  // 1/sqrt(64) * log2(e)
  const int kend = q0 + 128;
  const int trow = threadIdx.x >> 3;        // 0..31 (key row this thread stages)
  const int tc0  = (threadIdx.x & 7) * 8;   // 0..56 (d chunk this thread stages)

  for (int kt = 0; kt < kend; kt += 32) {
    // ---- stage K via async DMA, V via register transpose ----
    {
      async_copy16(Kp + (size_t)(kt + trow) * 64 + tc0, &Klds[trow][tc0]);
      v8bf vv = ld8(Vp + (size_t)(kt + trow) * 64 + tc0);
#pragma unroll
      for (int i = 0; i < 8; ++i) Vtlds[tc0 + i][trow] = vv[i];
      wait_async0();               // our K DMA landed in LDS
    }
    __syncthreads();               // everyone's K/V staged

    // ---- S = Q @ K^T  (two 16-key subtiles) ----
    v8f s0 = zero, s1 = zero;
    {
      const __bf16* p00 = &Klds[lnib][bkoff];
      const __bf16* p10 = &Klds[16 + lnib][bkoff];
      s0 = wmma_bf16(qa1, cat16(ld8(p00 + 32), ld8(p00 + 40)), s0);
      s0 = wmma_bf16(qa0, cat16(ld8(p00),      ld8(p00 + 8)),  s0);
      s1 = wmma_bf16(qa1, cat16(ld8(p10 + 32), ld8(p10 + 40)), s1);
      s1 = wmma_bf16(qa0, cat16(ld8(p10),      ld8(p10 + 8)),  s1);
    }

    // ---- causal mask + online softmax (row lives across 16 lanes) ----
    float p0[8], p1[8], corr[8];
#pragma unroll
    for (int r = 0; r < 8; ++r) {
      int qg = q0 + wid * 16 + r + 8 * hi;
      bool m0b = (kt + lnib)      > qg;
      bool m1b = (kt + 16 + lnib) > qg;
      float x0 = m0b ? -1e30f : s0[r] * scale2;
      float x1 = m1b ? -1e30f : s1[r] * scale2;
      float mx = fmaxf(x0, x1);
      mx = fmaxf(mx, __shfl_xor(mx, 1));
      mx = fmaxf(mx, __shfl_xor(mx, 2));
      mx = fmaxf(mx, __shfl_xor(mx, 4));
      mx = fmaxf(mx, __shfl_xor(mx, 8));
      float mnew = fmaxf(mrow[r], mx);
      float cr = exp2f(mrow[r] - mnew);
      mrow[r] = mnew;
      corr[r] = cr;
      p0[r] = m0b ? 0.f : exp2f(x0 - mnew);
      p1[r] = m1b ? 0.f : exp2f(x1 - mnew);
      float ps = p0[r] + p1[r];
      ps += __shfl_xor(ps, 1);
      ps += __shfl_xor(ps, 2);
      ps += __shfl_xor(ps, 4);
      ps += __shfl_xor(ps, 8);
      lrow[r] = lrow[r] * cr + ps;
    }

    // rescale running output
#pragma unroll
    for (int r = 0; r < 8; ++r) {
      o0[r] *= corr[r]; o1[r] *= corr[r]; o2[r] *= corr[r]; o3[r] *= corr[r];
    }

    // ---- C-layout -> A-layout for P via per-wave LDS region ----
#pragma unroll
    for (int r = 0; r < 8; ++r) {
      Plds[wid][r + 8 * hi][lnib]      = (__bf16)p0[r];
      Plds[wid][r + 8 * hi][lnib + 16] = (__bf16)p1[r];
    }
    const __bf16* pp = &Plds[wid][lnib][hi ? 16 : 0];
    v16bf pa = cat16(ld8(pp), ld8(pp + 8));

    // ---- O += P @ V  (V^T staged, B-frags contiguous) ----
    {
      const __bf16* v0p = &Vtlds[lnib][bkoff];
      o0 = wmma_bf16(pa, cat16(ld8(v0p), ld8(v0p + 8)), o0);
      const __bf16* v1p = &Vtlds[16 + lnib][bkoff];
      o1 = wmma_bf16(pa, cat16(ld8(v1p), ld8(v1p + 8)), o1);
      const __bf16* v2p = &Vtlds[32 + lnib][bkoff];
      o2 = wmma_bf16(pa, cat16(ld8(v2p), ld8(v2p + 8)), o2);
      const __bf16* v3p = &Vtlds[48 + lnib][bkoff];
      o3 = wmma_bf16(pa, cat16(ld8(v3p), ld8(v3p + 8)), o3);
    }
    __syncthreads();
  }

  // ---- normalize and store to attn-out [B,T,C] bf16 ----
  const int b = bh >> 3, h = bh & 7;
#pragma unroll
  for (int r = 0; r < 8; ++r) {
    float inv = 1.0f / lrow[r];
    int t = q0 + wid * 16 + r + 8 * hi;
    size_t base = ((size_t)(b * 2048 + t)) * 512 + h * 64 + lnib;
    ao[base +  0] = (__bf16)(o0[r] * inv);
    ao[base + 16] = (__bf16)(o1[r] * inv);
    ao[base + 32] = (__bf16)(o2[r] * inv);
    ao[base + 48] = (__bf16)(o3[r] * inv);
  }
}

// ---------------------------------------------------------------------------
// Output projection: ao[8192,512] @ WoutT -> f32 out [8192,512]
// ---------------------------------------------------------------------------
__global__ __launch_bounds__(256) void out_gemm_kernel(
    const __bf16* __restrict__ ao, const __bf16* __restrict__ woutT,
    float* __restrict__ out) {
  const int lane = threadIdx.x & 31, wid = threadIdx.x >> 5;
  const int m0 = blockIdx.x * 128 + wid * 16;
  const int n0 = blockIdx.y * 64;
  const v8f zero = {0.f, 0.f, 0.f, 0.f, 0.f, 0.f, 0.f, 0.f};
  v8f acc[4] = {zero, zero, zero, zero};
  gemm_wave_acc(ao, woutT, 512, m0, n0, lane, acc);

  const int hi = lane >> 4, lnib = lane & 15;
#pragma unroll
  for (int j = 0; j < 4; ++j) {
#pragma unroll
    for (int r = 0; r < 8; ++r) {
      int m = m0 + r + 8 * hi;
      int n = n0 + j * 16 + lnib;
      out[(size_t)m * 512 + n] = acc[j][r];
    }
  }
}

// ---------------------------------------------------------------------------
// Launch
// ---------------------------------------------------------------------------
extern "C" void kernel_launch(void* const* d_in, const int* in_sizes, int n_in,
                              void* d_out, int out_size, void* d_ws, size_t ws_size,
                              hipStream_t stream) {
  const float* x    = (const float*)d_in[0];  // [4,2048,512]
  const float* Wqkv = (const float*)d_in[1];  // [512,1536]
  const float* Wout = (const float*)d_in[2];  // [512,512]
  float* out = (float*)d_out;

  char* ws = (char*)d_ws;
  const size_t XB_BYTES = (size_t)8192 * 512 * 2;      // 8 MB
  const size_t WQ_BYTES = (size_t)1536 * 512 * 2;      // 1.5 MB
  const size_t WO_BYTES = (size_t)512 * 512 * 2;       // 0.5 MB
  const size_t HEADBUF  = (size_t)32 * 2048 * 64 * 2;  // 8 MB each

  __bf16* xb    = (__bf16*)(ws);
  __bf16* wqkvT = (__bf16*)(ws + XB_BYTES);
  __bf16* woutT = (__bf16*)(ws + XB_BYTES + WQ_BYTES);
  __bf16* qb    = (__bf16*)(ws + XB_BYTES + WQ_BYTES + WO_BYTES);
  __bf16* kbuf  = (__bf16*)(ws + XB_BYTES + WQ_BYTES + WO_BYTES + HEADBUF);
  __bf16* vbuf  = (__bf16*)(ws + XB_BYTES + WQ_BYTES + WO_BYTES + 2 * HEADBUF);
  __bf16* ao    = (__bf16*)(ws + XB_BYTES + WQ_BYTES + WO_BYTES + 3 * HEADBUF);

  // 1. precision conversion + weight transposition
  conv_f32_to_bf16<<<(8192 * 512 + 255) / 256, 256, 0, stream>>>(x, xb, 8192 * 512);
  conv_transpose_bf16<<<(512 * 1536 + 255) / 256, 256, 0, stream>>>(Wqkv, wqkvT, 512, 1536);
  conv_transpose_bf16<<<(512 * 512 + 255) / 256, 256, 0, stream>>>(Wout, woutT, 512, 512);

  // 2. QKV projection (scatters into [B,H,T,D])
  qkv_gemm_kernel<<<dim3(8192 / 128, 1536 / 64), 256, 0, stream>>>(xb, wqkvT, qb, kbuf, vbuf);

  // 3. causal flash attention
  attn_kernel<<<dim3(32, 2048 / 128), 256, 0, stream>>>(qb, kbuf, vbuf, ao);

  // 4. output projection -> f32
  out_gemm_kernel<<<dim3(8192 / 128, 512 / 64), 256, 0, stream>>>(ao, woutT, out);
}
